// AIRSSpectralGNN_49606872268973
// MI455X (gfx1250) — matrix operand order repaired and checked by reading, not compiled, not run on hardware
//
#include <hip/hip_runtime.h>

#define DH 128
#define NEG_SLOPE 0.2f
#define SM_EPS 1e-16f
#define LN_EPS 1e-5f

typedef __attribute__((ext_vector_type(16))) __bf16 v16bf;
typedef __attribute__((ext_vector_type(8)))  float  v8f;

static __device__ __forceinline__ unsigned short f2bf(float f) {
  unsigned int u = __float_as_uint(f);
  unsigned int r = (u + 0x7FFFu + ((u >> 16) & 1u)) >> 16;  // RNE
  return (unsigned short)r;
}

static __device__ __forceinline__ void atomicMaxF(float* addr, float v) {
  // works for mixed signs when memory is initialized to -inf
  if (v >= 0.f) atomicMax((int*)addr, __float_as_int(v));
  else          atomicMin((unsigned int*)addr, (unsigned int)__float_as_int(v));
}

// ---------------- generic fill ----------------
__global__ void k_fill(float* __restrict__ p, size_t n, float v) {
  size_t i = (size_t)blockIdx.x * blockDim.x + threadIdx.x;
  if (i < n) p[i] = v;
}

// ---------------- degree + self-loop edge_attr (mean of incoming) ----------------
__global__ void k_deg(const int* __restrict__ dst, const float* __restrict__ eattr,
                      float* __restrict__ deg, float* __restrict__ lsum, int E) {
  int e = blockIdx.x * blockDim.x + threadIdx.x;
  if (e >= E) return;
  int d = dst[e];
  atomicAdd(deg + d, 1.0f);
  atomicAdd(lsum + 3 * (size_t)d + 0, eattr[3 * (size_t)e + 0]);
  atomicAdd(lsum + 3 * (size_t)d + 1, eattr[3 * (size_t)e + 1]);
  atomicAdd(lsum + 3 * (size_t)d + 2, eattr[3 * (size_t)e + 2]);
}

__global__ void k_loopdiv(float* __restrict__ lsum, const float* __restrict__ deg, int N) {
  int i = blockIdx.x * blockDim.x + threadIdx.x;
  if (i >= 3 * N) return;
  lsum[i] = lsum[i] / fmaxf(deg[i / 3], 1.0f);
}

// ---------------- f32 -> bf16 convert (activations) ----------------
__global__ void k_tobf(const float* __restrict__ x, unsigned short* __restrict__ hb, size_t n) {
  size_t i = (size_t)blockIdx.x * blockDim.x + threadIdx.x;
  if (i < n) hb[i] = f2bf(x[i]);
}

// ---------------- w3 = W_e @ att_edge  (3 floats) ----------------
__global__ void k_w3(const float* __restrict__ We, const float* __restrict__ att_e,
                     float* __restrict__ w3) {
  int j = threadIdx.x >> 5;     // 3 waves, one per output
  int lane = threadIdx.x & 31;
  float s = 0.f;
  for (int k = lane; k < DH; k += 32) s += We[j * DH + k] * att_e[k];
  for (int o = 16; o; o >>= 1) s += __shfl_xor(s, o, 32);
  if (lane == 0) w3[j] = s;
}

// ---------------- pack W [din,128] f32 -> B-fragment bf16 layout ----------------
// fragment index: ((colTile*ksteps + ks)*32 + lane) -> 8 dwords (16 bf16)
// dword v: lo half = W[k0 + 2v + klaneoff][col], hi half = W[k0 + 2v + 1 + klaneoff][col]
// klaneoff = 16 for lanes 16..31, col = colTile*16 + (lane&15)
__global__ void k_pack_w(const float* __restrict__ W, uint4* __restrict__ wf, int din) {
  int ksteps = din >> 5;
  int tid = blockIdx.x * blockDim.x + threadIdx.x;
  if (tid >= 8 * ksteps * 32) return;
  int lane = tid & 31;
  int rest = tid >> 5;
  int ks = rest % ksteps;
  int c = rest / ksteps;
  int col = (c << 4) + (lane & 15);
  int kbase = (ks << 5) + ((lane >> 4) << 4);
  unsigned int d[8];
#pragma unroll
  for (int v = 0; v < 8; ++v) {
    unsigned int lo = f2bf(W[(size_t)(kbase + 2 * v) * DH + col]);
    unsigned int hi = f2bf(W[(size_t)(kbase + 2 * v + 1) * DH + col]);
    d[v] = lo | (hi << 16);
  }
  wf[2 * (size_t)tid]     = make_uint4(d[0], d[1], d[2], d[3]);
  wf[2 * (size_t)tid + 1] = make_uint4(d[4], d[5], d[6], d[7]);
}

// ---------------- hp = h(bf16) @ W(frag)  via V_WMMA_F32_16X16X32_BF16 ----------------
// block = 256 threads = 8 waves; wave w -> column tile w (16 cols), block row-tile = 16 nodes
__global__ __launch_bounds__(256)
void k_gemm(const unsigned short* __restrict__ hbf, const uint4* __restrict__ wf,
            float* __restrict__ hp, int din) {
  int lane = threadIdx.x & 31;
  int wave = threadIdx.x >> 5;          // col tile 0..7
  int rowTile = blockIdx.x << 4;
  int r = lane & 15;
  int hi = lane >> 4;
  int ksteps = din >> 5;
  const unsigned short* arow = hbf + (size_t)(rowTile + r) * din;
  v8f acc = {};
  for (int ks = 0; ks < ksteps; ++ks) {
    int k0 = ks << 5;
    union { uint4 q[2]; v16bf v; } A, B;
    // 16-bit A 16x32 layout (matches packed B): lanes<16 hold K {0..15}, lanes>=16 hold K {16..31}
    A.q[0] = *(const uint4*)(arow + k0 + hi * 16);
    A.q[1] = *(const uint4*)(arow + k0 + hi * 16 + 8);
    const uint4* bp = wf + (((size_t)(wave * ksteps + ks) << 5) + lane) * 2;
    B.q[0] = bp[0];
    B.q[1] = bp[1];
    acc = __builtin_amdgcn_wmma_f32_16x16x32_bf16(false, A.v, false, B.v,
                                                  (short)0, acc, false, false);
  }
  // C/D layout: VGPR m -> (M = m + 8*hi, N = lane&15)
  float* op = hp + (size_t)rowTile * DH + (wave << 4) + r;
#pragma unroll
  for (int m = 0; m < 8; ++m) op[(size_t)(m + hi * 8) * DH] = acc[m];
}

// ---------------- a_src/a_dst = hp . att_{src,dst} (wave per node) ----------------
__global__ void k_attn(const float* __restrict__ hp, const float* __restrict__ asv,
                       const float* __restrict__ adv, float* __restrict__ a_src,
                       float* __restrict__ a_dst, int N) {
  int gid = blockIdx.x * blockDim.x + threadIdx.x;
  int node = gid >> 5, lane = gid & 31;
  if (node >= N) return;
  const float* row = hp + (size_t)node * DH;
  float s0 = 0.f, s1 = 0.f;
#pragma unroll
  for (int t = 0; t < 4; ++t) {
    int j = lane + 32 * t;
    float h = row[j];
    s0 += h * asv[j];
    s1 += h * adv[j];
  }
  for (int o = 16; o; o >>= 1) { s0 += __shfl_xor(s0, o, 32); s1 += __shfl_xor(s1, o, 32); }
  if (lane == 0) { a_src[node] = s0; a_dst[node] = s1; }
}

// ---------------- edge pass 1: raw alpha + segment max ----------------
__global__ void k_edge1(const int* __restrict__ src, const int* __restrict__ dst,
                        const float* __restrict__ eattr, const float* __restrict__ lattr,
                        const float* __restrict__ w3, const float* __restrict__ a_src,
                        const float* __restrict__ a_dst, float* __restrict__ alpha,
                        float* __restrict__ mmax, int E, int N) {
  int e = blockIdx.x * blockDim.x + threadIdx.x;
  if (e >= E + N) return;
  int s, d; const float* ap;
  if (e < E) { s = src[e]; d = dst[e]; ap = eattr + 3 * (size_t)e; }
  else       { s = d = e - E;          ap = lattr + 3 * (size_t)(e - E); }
  float ae = ap[0] * w3[0] + ap[1] * w3[1] + ap[2] * w3[2];
  float a = a_src[s] + a_dst[d] + ae;
  a = a > 0.f ? a : NEG_SLOPE * a;
  alpha[e] = a;
  atomicMaxF(mmax + d, a);
}

// ---------------- edge pass 2: exp + segment sum ----------------
__global__ void k_edge2(const int* __restrict__ dst, float* __restrict__ alpha,
                        const float* __restrict__ mmax, float* __restrict__ denom,
                        int E, int N) {
  int e = blockIdx.x * blockDim.x + threadIdx.x;
  if (e >= E + N) return;
  int d = (e < E) ? dst[e] : (e - E);
  float a = expf(alpha[e] - mmax[d]);
  alpha[e] = a;
  atomicAdd(denom + d, a);
}

// ---------------- edge pass 3: aggregate (wave per edge, float4 per lane) ----------------
__global__ void k_edge3(const int* __restrict__ src, const int* __restrict__ dst,
                        const float* __restrict__ alpha, const float* __restrict__ denom,
                        const float* __restrict__ hp, float* __restrict__ out, int E, int N) {
  int gid = blockIdx.x * blockDim.x + threadIdx.x;
  int e = gid >> 5, lane = gid & 31;
  if (e >= E + N) return;
  int s, d;
  if (e < E) { s = src[e]; d = dst[e]; } else { s = d = e - E; }
  float coef = alpha[e] / (denom[d] + SM_EPS);
  const float4 hv = *(const float4*)(hp + (size_t)s * DH + lane * 4);
  float* ob = out + (size_t)d * DH + lane * 4;
  atomicAdd(ob + 0, coef * hv.x);
  atomicAdd(ob + 1, coef * hv.y);
  atomicAdd(ob + 2, coef * hv.z);
  atomicAdd(ob + 3, coef * hv.w);
}

// ---------------- finalize (layers 0..2): relu(out+bias) -> bf16 next-layer input ----------------
__global__ void k_finalize(const float* __restrict__ out, const float* __restrict__ bias,
                           unsigned short* __restrict__ hbf, size_t n) {
  size_t i = (size_t)blockIdx.x * blockDim.x + threadIdx.x;
  if (i >= n) return;
  float v = out[i] + bias[i & (DH - 1)];
  v = v > 0.f ? v : 0.f;
  hbf[i] = f2bf(v);
}

// ---------------- final layer: relu(out+bias) then LayerNorm (wave per node) ----------------
__global__ void k_ln(const float* __restrict__ out, const float* __restrict__ bias,
                     const float* __restrict__ g, const float* __restrict__ b,
                     float* __restrict__ y, int N) {
  int gid = blockIdx.x * blockDim.x + threadIdx.x;
  int node = gid >> 5, lane = gid & 31;
  if (node >= N) return;
  const float* row = out + (size_t)node * DH;
  float v[4]; float s = 0.f;
#pragma unroll
  for (int t = 0; t < 4; ++t) {
    int j = lane + 32 * t;
    float x = row[j] + bias[j];
    x = x > 0.f ? x : 0.f;
    v[t] = x; s += x;
  }
  for (int o = 16; o; o >>= 1) s += __shfl_xor(s, o, 32);
  float mu = s * (1.0f / DH);
  float q = 0.f;
#pragma unroll
  for (int t = 0; t < 4; ++t) { float dd = v[t] - mu; q += dd * dd; }
  for (int o = 16; o; o >>= 1) q += __shfl_xor(q, o, 32);
  float inv = rsqrtf(q * (1.0f / DH) + LN_EPS);
#pragma unroll
  for (int t = 0; t < 4; ++t) {
    int j = lane + 32 * t;
    y[(size_t)node * DH + j] = (v[t] - mu) * inv * g[j] + b[j];
  }
}

extern "C" void kernel_launch(void* const* d_in, const int* in_sizes, int n_in,
                              void* d_out, int out_size, void* d_ws, size_t ws_size,
                              hipStream_t stream) {
  const int N = in_sizes[0] / 32;        // 50000
  const int E = in_sizes[1] / 2;         // 800000
  const float* x     = (const float*)d_in[0];
  const int*   ei    = (const int*)d_in[1];
  const float* eattr = (const float*)d_in[2];
  const int* src = ei;
  const int* dst = ei + E;
  // params flattened in insertion order: per layer {W, W_e, att_src, att_dst, att_edge, bias}, then ln_g, ln_b
  const float* ln_g = (const float*)d_in[3 + 6 * 4 + 0];
  const float* ln_b = (const float*)d_in[3 + 6 * 4 + 1];

  // ---- workspace carve-out ----
  char* p = (char*)d_ws;
  auto alloc = [&](size_t bytes) -> void* {
    void* r = (void*)p;
    p += (bytes + 255) & ~(size_t)255;
    return r;
  };
  unsigned short* hbf  = (unsigned short*)alloc((size_t)N * DH * 2);
  float* hp    = (float*)alloc((size_t)N * DH * 4);
  float* outb  = (float*)alloc((size_t)N * DH * 4);
  float* a_src = (float*)alloc((size_t)N * 4);
  float* a_dst = (float*)alloc((size_t)N * 4);
  float* mmax  = (float*)alloc((size_t)N * 4);
  float* denom = (float*)alloc((size_t)N * 4);
  float* deg   = (float*)alloc((size_t)N * 4);
  float* lattr = (float*)alloc((size_t)N * 3 * 4);
  float* alpha = (float*)alloc((size_t)(E + N) * 4);
  uint4* wfrag = (uint4*)alloc((size_t)8 * 4 * 32 * 32);
  float* w3    = (float*)alloc(64);

  const int T = 256;
  auto blk = [](size_t n, int t) { return (unsigned)((n + t - 1) / t); };

  // ---- prep: degree + mean self-loop attr, x -> bf16 ----
  k_fill<<<blk((size_t)N, T), T, 0, stream>>>(deg, (size_t)N, 0.f);
  k_fill<<<blk((size_t)N * 3, T), T, 0, stream>>>(lattr, (size_t)N * 3, 0.f);
  k_deg<<<blk((size_t)E, T), T, 0, stream>>>(dst, eattr, deg, lattr, E);
  k_loopdiv<<<blk((size_t)N * 3, T), T, 0, stream>>>(lattr, deg, N);
  k_tobf<<<blk((size_t)N * 32, T), T, 0, stream>>>(x, hbf, (size_t)N * 32);

  const size_t EN = (size_t)E + N;
  for (int l = 0; l < 4; ++l) {
    const float* W    = (const float*)d_in[3 + 6 * l + 0];
    const float* W_e  = (const float*)d_in[3 + 6 * l + 1];
    const float* asv  = (const float*)d_in[3 + 6 * l + 2];
    const float* adv  = (const float*)d_in[3 + 6 * l + 3];
    const float* aev  = (const float*)d_in[3 + 6 * l + 4];
    const float* bias = (const float*)d_in[3 + 6 * l + 5];
    const int din = (l == 0) ? 32 : DH;
    const int ksteps = din >> 5;

    k_w3<<<1, 96, 0, stream>>>(W_e, aev, w3);
    k_pack_w<<<blk((size_t)8 * ksteps * 32, T), T, 0, stream>>>(W, wfrag, din);
    k_fill<<<blk((size_t)N, T), T, 0, stream>>>(mmax, (size_t)N, -INFINITY);
    k_fill<<<blk((size_t)N, T), T, 0, stream>>>(denom, (size_t)N, 0.f);
    k_fill<<<blk((size_t)N * DH, T), T, 0, stream>>>(outb, (size_t)N * DH, 0.f);

    k_gemm<<<N / 16, T, 0, stream>>>(hbf, wfrag, hp, din);
    k_attn<<<blk((size_t)N * 32, T), T, 0, stream>>>(hp, asv, adv, a_src, a_dst, N);
    k_edge1<<<blk(EN, T), T, 0, stream>>>(src, dst, eattr, lattr, w3, a_src, a_dst,
                                          alpha, mmax, E, N);
    k_edge2<<<blk(EN, T), T, 0, stream>>>(dst, alpha, mmax, denom, E, N);
    k_edge3<<<blk(EN * 32, T), T, 0, stream>>>(src, dst, alpha, denom, hp, outb, E, N);

    if (l < 3) {
      k_finalize<<<blk((size_t)N * DH, T), T, 0, stream>>>(outb, bias, hbf, (size_t)N * DH);
    } else {
      k_ln<<<blk((size_t)N * 32, T), T, 0, stream>>>(outb, bias, ln_g, ln_b,
                                                     (float*)d_out, N);
    }
  }
}
